// EGConv_layer_72258529788101
// MI455X (gfx1250) — matrix operand order repaired and checked by reading, not compiled, not run on hardware
//
#include <hip/hip_runtime.h>

#define NNODES 100000
#define NEDGES 3200000
#define DDIM   128
#define NGRAPH 64
#define GN_EPS 1e-5f

typedef __attribute__((ext_vector_type(16))) __bf16 v16bf;
typedef __attribute__((ext_vector_type(8)))  float  v8f;

// monotonic float<->uint mapping: integer max == float max
__device__ __forceinline__ unsigned fmap(float f) {
  union { float f; unsigned u; } v; v.f = f;
  return (v.u & 0x80000000u) ? ~v.u : (v.u | 0x80000000u);
}
__device__ __forceinline__ float funmap(unsigned m) {
  union { unsigned u; float f; } v;
  v.u = (m & 0x80000000u) ? (m & 0x7FFFFFFFu) : ~m;
  return v.f;
}

// ---- prep: W = [W_bases | W_comb] as bf16, stored column-major (K contiguous) ----
__global__ void prep_weights(const float* __restrict__ Wb, const float* __restrict__ Wc,
                             __bf16* __restrict__ Wbf) {
  int t = blockIdx.x * 256 + threadIdx.x;
  if (t >= DDIM * DDIM) return;
  int col = t >> 7, k = t & 127;
  float v = (col < 64) ? Wb[k * 64 + col] : Wc[k * 64 + (col - 64)];
  Wbf[col * DDIM + k] = (__bf16)v;   // native v_cvt bf16
}

__global__ void zero_stats(float* __restrict__ p, int n) {
  int t = blockIdx.x * 256 + threadIdx.x;
  if (t < n) p[t] = 0.0f;
}

// ---- fused GEMM: P[N,128] = node[N,128] @ W[128,128] (+ b_comb on cols 64..127) ----
__global__ void gemm_wmma(const float* __restrict__ node,
                          const unsigned* __restrict__ wU,   // bf16 pairs, col-major
                          const float* __restrict__ b_comb,
                          float* __restrict__ P) {
  const int wave = threadIdx.x >> 5;
  const int lane = threadIdx.x & 31;
  const int tile = blockIdx.x * 8 + wave;          // 16-row tile, wave-uniform
  if (tile >= NNODES / 16) return;
  const int row0 = tile * 16;
  const int half = lane >> 4;
  const int l16  = lane & 15;
  const float* arp = node + (size_t)(row0 + l16) * DDIM;

  v8f acc[8];
  #pragma unroll
  for (int ct = 0; ct < 8; ct++) acc[ct] = (v8f){0,0,0,0,0,0,0,0};

  union AV { v16bf v; __bf16 b[16]; };
  union BV { v16bf v; unsigned u[8]; };

  #pragma unroll
  for (int ks = 0; ks < 4; ks++) {
    const int k0 = ks * 32;
    AV av;                                  // ISA 16-bit A layout: vj -> K = sel + half*8 + 2(j%4)
    #pragma unroll
    for (int j = 0; j < 8; j++) {
      int kk = k0 + ((j < 4) ? 0 : 16) + half * 8 + 2 * (j & 3);
      float2 f = *(const float2*)(arp + kk);
      av.b[2 * j]     = (__bf16)f.x;        // compiler: v_cvt_pk_bf16_f32
      av.b[2 * j + 1] = (__bf16)f.y;
    }
    #pragma unroll
    for (int ct = 0; ct < 8; ct++) {        // B layout matches A: vj -> K = sel + half*8 + 2(j%4)
      const int col = ct * 16 + l16;
      const unsigned* bp = wU + (size_t)col * 64 + ((k0 + half * 8) >> 1);
      BV bv;
      #pragma unroll
      for (int j = 0; j < 8; j++) bv.u[j] = bp[(j & 3) + ((j >> 2) << 3)];
      acc[ct] = __builtin_amdgcn_wmma_f32_16x16x32_bf16(
          false, av.v, false, bv.v, (short)0, acc[ct], false, false);
    }
  }
  #pragma unroll
  for (int ct = 0; ct < 8; ct++) {          // C/D layout: reg r -> M = r + 8*half
    const int col = ct * 16 + l16;
    const float bc = (col >= 64) ? b_comb[col - 64] : 0.0f;
    #pragma unroll
    for (int r = 0; r < 8; r++)
      P[(size_t)(row0 + r + 8 * half) * DDIM + col] = acc[ct][r] + bc;
  }
}

// ---- self-loop init: agg_sum = agg_max = bases ----
__global__ void init_agg(const float* __restrict__ P, float* __restrict__ aggS,
                         unsigned* __restrict__ aggM) {
  int t = blockIdx.x * 256 + threadIdx.x;
  if (t >= NNODES * 64) return;
  int n = t >> 6, c = t & 63;
  float b = P[(size_t)n * DDIM + c];
  aggS[t] = b;
  aggM[t] = fmap(b);
}

// ---- edge scatter: 16 threads/edge, float4 per thread, native atomics ----
__global__ void edge_scatter(const int* __restrict__ ei, const float* __restrict__ P,
                             float* __restrict__ aggS, unsigned* __restrict__ aggM) {
  int t = blockIdx.x * 256 + threadIdx.x;
  if (t >= NEDGES * 16) return;
  int e = t >> 4, part = t & 15;
  int src = ei[e], dst = ei[NEDGES + e];
  const float4 m = *(const float4*)(P + (size_t)src * DDIM + part * 4);
  float*    sp = aggS + (size_t)dst * 64 + part * 4;
  unsigned* mp = aggM + (size_t)dst * 64 + part * 4;
  atomicAdd(sp + 0, m.x); atomicAdd(sp + 1, m.y);
  atomicAdd(sp + 2, m.z); atomicAdd(sp + 3, m.w);
  atomicMax(mp + 0, fmap(m.x)); atomicMax(mp + 1, fmap(m.y));
  atomicMax(mp + 2, fmap(m.z)); atomicMax(mp + 3, fmap(m.w));
}

// ---- per-node combine (comb[8x8] @ aggr[8x16]) + bias, + graph mean accumulation ----
__global__ void combine(const float* __restrict__ P, const float* __restrict__ aggS,
                        const unsigned* __restrict__ aggM, const float* __restrict__ bias_out,
                        const int* __restrict__ batch, float* __restrict__ hbuf,
                        float* __restrict__ sum_h, float* __restrict__ cnt) {
  const int wave = threadIdx.x >> 5;
  const int lane = threadIdx.x & 31;
  const int n = blockIdx.x * 8 + wave;             // one wave per node
  if (n >= NNODES) return;
  const int h  = lane >> 2;
  const int f0 = (lane & 3) * 4;
  const float* cp = P + (size_t)n * DDIM + 64 + h * 8;
  float c[8];
  #pragma unroll
  for (int k = 0; k < 8; k++) c[k] = cp[k];
  float o[4] = {0, 0, 0, 0};
  const float*    sp = aggS + (size_t)n * 64;
  const unsigned* mp = aggM + (size_t)n * 64;
  #pragma unroll
  for (int k = 0; k < 4; k++) {                    // k=0..3 -> sum aggregator
    const float4 s = *(const float4*)(sp + k * 16 + f0);
    o[0] += c[k] * s.x; o[1] += c[k] * s.y; o[2] += c[k] * s.z; o[3] += c[k] * s.w;
  }
  #pragma unroll
  for (int k = 0; k < 4; k++) {                    // k=4..7 -> max aggregator
    const uint4 mu = *(const uint4*)(mp + k * 16 + f0);
    o[0] += c[4 + k] * funmap(mu.x); o[1] += c[4 + k] * funmap(mu.y);
    o[2] += c[4 + k] * funmap(mu.z); o[3] += c[4 + k] * funmap(mu.w);
  }
  const int g = batch[n];
  #pragma unroll
  for (int j = 0; j < 4; j++) {
    int d = h * 16 + f0 + j;
    float val = o[j] + bias_out[d];
    hbuf[(size_t)n * DDIM + d] = val;
    atomicAdd(&sum_h[g * DDIM + d], val);
  }
  if (lane == 0) atomicAdd(&cnt[g], 1.0f);
}

__global__ void finalize_mean(const float* __restrict__ sum_h, const float* __restrict__ cnt,
                              float* __restrict__ mean) {
  int t = blockIdx.x * 256 + threadIdx.x;
  if (t >= NGRAPH * DDIM) return;
  mean[t] = sum_h[t] / fmaxf(cnt[t >> 7], 1.0f);
}

__global__ void var_accum(const float* __restrict__ hbuf, const float* __restrict__ mean,
                          const float* __restrict__ mean_scale, const int* __restrict__ batch,
                          float* __restrict__ var_sum) {
  int t = blockIdx.x * 256 + threadIdx.x;
  if (t >= NNODES * DDIM) return;
  int n = t >> 7, d = t & 127;
  int g = batch[n];
  float xc = hbuf[t] - mean[g * DDIM + d] * mean_scale[d];
  atomicAdd(&var_sum[g * DDIM + d], xc * xc);
}

__global__ void finalize_std(const float* __restrict__ var_sum, const float* __restrict__ cnt,
                             float* __restrict__ inv_std) {
  int t = blockIdx.x * 256 + threadIdx.x;
  if (t >= NGRAPH * DDIM) return;
  inv_std[t] = rsqrtf(var_sum[t] / fmaxf(cnt[t >> 7], 1.0f) + GN_EPS);
}

__global__ void norm_out(const float* __restrict__ hbuf, const float* __restrict__ mean,
                         const float* __restrict__ inv_std, const float* __restrict__ mean_scale,
                         const float* __restrict__ w, const float* __restrict__ b,
                         const int* __restrict__ batch, float* __restrict__ out) {
  int t = blockIdx.x * 256 + threadIdx.x;
  if (t >= NNODES * DDIM) return;
  int n = t >> 7, d = t & 127;
  int g = batch[n];
  float xc = hbuf[t] - mean[g * DDIM + d] * mean_scale[d];
  float y = w[d] * xc * inv_std[g * DDIM + d] + b[d];
  out[t] = fmaxf(y, 0.0f);
}

extern "C" void kernel_launch(void* const* d_in, const int* in_sizes, int n_in,
                              void* d_out, int out_size, void* d_ws, size_t ws_size,
                              hipStream_t stream) {
  const float* node    = (const float*)d_in[0];
  const int*   ei      = (const int*)  d_in[1];
  const int*   batch   = (const int*)  d_in[3];
  const float* W_bases = (const float*)d_in[4];
  const float* W_comb  = (const float*)d_in[5];
  const float* b_comb  = (const float*)d_in[6];
  const float* bias_o  = (const float*)d_in[7];
  const float* gn_w    = (const float*)d_in[8];
  const float* gn_b    = (const float*)d_in[9];
  const float* gn_ms   = (const float*)d_in[10];
  float* out = (float*)d_out;

  // workspace layout
  float*    ws      = (float*)d_ws;
  float*    P       = ws;                                   // N*128
  float*    aggS    = P    + (size_t)NNODES * DDIM;         // N*64
  unsigned* aggM    = (unsigned*)(aggS + (size_t)NNODES * 64);  // N*64
  float*    hbuf    = (float*)(aggM + (size_t)NNODES * 64); // N*128
  float*    sum_h   = hbuf + (size_t)NNODES * DDIM;         // G*128
  float*    cnt     = sum_h + NGRAPH * DDIM;                // G
  float*    var_sum = cnt + NGRAPH;                         // G*128
  float*    mean    = var_sum + NGRAPH * DDIM;              // G*128
  float*    inv_std = mean + NGRAPH * DDIM;                 // G*128
  __bf16*   Wbf     = (__bf16*)(inv_std + NGRAPH * DDIM);   // 128*128 bf16

  const int zeroN = NGRAPH * DDIM + NGRAPH + NGRAPH * DDIM; // sum_h, cnt, var_sum
  prep_weights<<<(DDIM * DDIM + 255) / 256, 256, 0, stream>>>(W_bases, W_comb, Wbf);
  zero_stats  <<<(zeroN + 255) / 256,       256, 0, stream>>>(sum_h, zeroN);
  gemm_wmma   <<<(NNODES / 16 + 7) / 8,     256, 0, stream>>>(node, (const unsigned*)Wbf, b_comb, P);
  init_agg    <<<(NNODES * 64 + 255) / 256, 256, 0, stream>>>(P, aggS, aggM);
  edge_scatter<<<(NEDGES * 16 + 255) / 256, 256, 0, stream>>>(ei, P, aggS, aggM);
  combine     <<<(NNODES + 7) / 8,          256, 0, stream>>>(P, aggS, aggM, bias_o, batch,
                                                              hbuf, sum_h, cnt);
  finalize_mean<<<(NGRAPH * DDIM + 255) / 256, 256, 0, stream>>>(sum_h, cnt, mean);
  var_accum   <<<(NNODES * DDIM + 255) / 256, 256, 0, stream>>>(hbuf, mean, gn_ms, batch, var_sum);
  finalize_std<<<(NGRAPH * DDIM + 255) / 256, 256, 0, stream>>>(var_sum, cnt, inv_std);
  norm_out    <<<(NNODES * DDIM + 255) / 256, 256, 0, stream>>>(hbuf, mean, inv_std, gn_ms,
                                                                gn_w, gn_b, batch, out);
}
